// EdgeGATConv_19799799234983
// MI455X (gfx1250) — matrix operand order repaired and checked by reading, not compiled, not run on hardware
//
#include <hip/hip_runtime.h>

// ---------------------------------------------------------------------------
// EdgeGATConv for MI455X (gfx1250), wave32 + WMMA.
//   h   = x @ W_fc + b_fc                       (fp32 WMMA 16x16x4)
//   a   = leaky_relu([h_i | h_j | eattr] @ W_att + b_att)  (bf16 WMMA 16x16x32)
//   out = scatter_add(h_j * a, row)             (f32 global atomics)
// ---------------------------------------------------------------------------

#define NNODES   100000
#define NEDGES   1000000
#define HID      128
#define EDIM     16
#define NEG_SLOPE 0.2f
#define KT_ATT   9          // K = 272 padded to 288 = 9 * 32
#define WPB      4          // waves per block in edge kernel

typedef __attribute__((ext_vector_type(2)))  float  v2f;
typedef __attribute__((ext_vector_type(8)))  float  v8f;
typedef __attribute__((ext_vector_type(16))) __bf16 v16bf;

union F32x2   { v2f   v; float          f[2];  };
union BF16x16 { v16bf v; unsigned short u[16]; uint4 q[2]; };

__device__ __forceinline__ unsigned short f32_to_bf16(float f) {
  unsigned int x = __float_as_uint(f);
  unsigned int r = x + 0x7FFFu + ((x >> 16) & 1u);   // round-to-nearest-even
  return (unsigned short)(r >> 16);
}

// ---------------------------------------------------------------------------
// d_out zero init (harness poisons output; we accumulate with atomics)
// ---------------------------------------------------------------------------
__global__ void zero_f32(float* __restrict__ p, int n) {
  int i = blockIdx.x * blockDim.x + threadIdx.x;
  if (i < n) p[i] = 0.0f;
}

// ---------------------------------------------------------------------------
// Pack W_fc [128,128] f32 into fp32 WMMA B-fragment layout (B is 4x16 f32):
//   VGPR0: K=kb+0 (lanes 0-15), K=kb+2 (lanes 16-31); VGPR1: +1 / +3
//   frag[(nt*32 + k0)*32 + lane] -> 2 floats
// ---------------------------------------------------------------------------
__global__ void pack_wfc(const float* __restrict__ W, float* __restrict__ out) {
  int idx = blockIdx.x * blockDim.x + threadIdx.x;     // 8*32*32
  if (idx >= 8 * 32 * 32) return;
  int lane = idx & 31;
  int k0   = (idx >> 5) & 31;
  int nt   = idx >> 10;
  int n    = nt * 16 + (lane & 15);
  int kb   = k0 * 4 + ((lane >= 16) ? 2 : 0);
  out[idx * 2 + 0] = W[(kb + 0) * HID + n];
  out[idx * 2 + 1] = W[(kb + 1) * HID + n];
}

// ---------------------------------------------------------------------------
// Pack W_att [272,128] f32 -> bf16 WMMA B-fragment layout (B is 32x16 bf16):
//   lane half h holds K=ktile*32+8h+0..7 (e<8) and +16+8h+0..7 (e>=8), N = lane&15.
//   frag[((nt*9 + kt)*32 + lane)*16 + e],  K >= 272 zero-padded.
// ---------------------------------------------------------------------------
__global__ void pack_watt(const float* __restrict__ W, unsigned short* __restrict__ out) {
  int idx = blockIdx.x * blockDim.x + threadIdx.x;     // 8*9*32*16
  if (idx >= 8 * KT_ATT * 32 * 16) return;
  int e     = idx & 15;
  int lane  = (idx >> 4) & 31;
  int chunk = idx >> 9;
  int kt    = chunk % KT_ATT;
  int nt    = chunk / KT_ATT;
  int n     = nt * 16 + (lane & 15);
  int k     = kt * 32 + ((lane >= 16) ? 8 : 0) + ((e >= 8) ? 8 : 0) + e;
  float w   = (k < 2 * HID + EDIM) ? W[k * HID + n] : 0.0f;
  out[idx]  = f32_to_bf16(w);
}

// ---------------------------------------------------------------------------
// Node GEMM: h = x @ W_fc + b_fc, fp32 WMMA 16x16x4.
// One wave per 16-row tile, 8 N-tiles, 32 K-steps. 6250 tiles exactly.
// ---------------------------------------------------------------------------
__global__ void __launch_bounds__(256)
node_gemm(const float* __restrict__ x, const float* __restrict__ wfrag,
          const float* __restrict__ bias, float* __restrict__ h,
          unsigned short* __restrict__ hbf) {
  const int lane = threadIdx.x & 31;
  const int wave = threadIdx.x >> 5;
  const int tile = blockIdx.x * 8 + wave;
  if (tile >= NNODES / 16) return;                      // uniform per-wave exit

  const int row  = tile * 16 + (lane & 15);             // A: M = lane&15
  const int ksub = (lane >= 16) ? 2 : 0;                // A: K sub-offset

  v8f acc[8];
  #pragma unroll
  for (int nt = 0; nt < 8; ++nt)
    #pragma unroll
    for (int v = 0; v < 8; ++v) acc[nt][v] = 0.0f;

  const float* xrow = x + (size_t)row * HID;
  for (int k0 = 0; k0 < 32; ++k0) {
    F32x2 a;
    a.f[0] = xrow[k0 * 4 + ksub + 0];
    a.f[1] = xrow[k0 * 4 + ksub + 1];
    #pragma unroll
    for (int nt = 0; nt < 8; ++nt) {
      F32x2 b;
      const float* bp = wfrag + ((size_t)(nt * 32 + k0) * 32 + lane) * 2;
      b.f[0] = bp[0];
      b.f[1] = bp[1];
      acc[nt] = __builtin_amdgcn_wmma_f32_16x16x4_f32(
          false, a.v, false, b.v, (short)0, acc[nt], false, false);
    }
  }

  // C layout: VGPR v = row (tile*16 + v) lanes 0-15, (+8) lanes 16-31; col = nt*16 + lane&15
  const int rbase = tile * 16 + ((lane >= 16) ? 8 : 0);
  const int c0    = lane & 15;
  #pragma unroll
  for (int nt = 0; nt < 8; ++nt) {
    const int col = nt * 16 + c0;
    const float bv = bias[col];
    #pragma unroll
    for (int v = 0; v < 8; ++v) {
      float val = acc[nt][v] + bv;
      int r = rbase + v;
      h[(size_t)r * HID + col]   = val;
      hbf[(size_t)r * HID + col] = f32_to_bf16(val);
    }
  }
}

// ---------------------------------------------------------------------------
// Edge kernel: one wave per 16 edges. bf16 WMMA 16x16x32, 9 K-tiles x 8 N-tiles.
// alpha staged through LDS, then float4 message multiply + f32 global atomics.
// ---------------------------------------------------------------------------
__global__ void __launch_bounds__(32 * WPB)
edge_gat(const long long* __restrict__ ei,
         const float* __restrict__ eattr,
         const unsigned short* __restrict__ wattfrag,
         const float* __restrict__ batt,
         const float* __restrict__ h,
         const unsigned short* __restrict__ hbf,
         float* __restrict__ out) {
  __shared__ int   s_row[WPB][16];
  __shared__ int   s_col[WPB][16];
  __shared__ float s_alpha[WPB][16][HID];               // 32 KB

  const int lane  = threadIdx.x & 31;
  const int wave  = threadIdx.x >> 5;
  const int group = blockIdx.x * WPB + wave;            // 16-edge group
  if (group >= NEDGES / 16) return;                     // 62500 exact, uniform
  const int ebase = group * 16;
  const int m     = lane & 15;
  const int hi    = (lane >= 16) ? 1 : 0;

  if (lane < 16) {
    s_row[wave][lane] = (int)ei[ebase + lane];          // dst
    s_col[wave][lane] = (int)ei[NEDGES + ebase + lane]; // src
  }
  // prefetch next groups' indices into caches (speculative)
  __builtin_prefetch(ei + ebase + 16 * WPB * 16, 0, 0);

  v8f acc[8];
  #pragma unroll
  for (int nt = 0; nt < 8; ++nt)
    #pragma unroll
    for (int v = 0; v < 8; ++v) acc[nt][v] = 0.0f;

  #pragma unroll
  for (int kt = 0; kt < KT_ATT; ++kt) {
    // ---- A fragment: 16x32 bf16, lane holds row m, two 16B chunks ----
    BF16x16 a;
    if (kt < 8) {
      const int node = (kt < 4) ? s_row[wave][m] : s_col[wave][m];
      const int kb   = (kt & 3) * 32 + (hi ? 8 : 0);
      const unsigned short* src = hbf + (size_t)node * HID + kb;
      a.q[0] = *(const uint4*)(src);                    // K = kb .. kb+7
      a.q[1] = *(const uint4*)(src + 16);               // K = kb+16 .. kb+23
    } else {
      // edge_attr tile: K 256..271 real (f32 -> bf16), K 272..287 zero pad
      const float* ep = eattr + (size_t)(ebase + m) * EDIM + (hi ? 8 : 0);
      #pragma unroll
      for (int e = 0; e < 8; ++e) a.u[e] = f32_to_bf16(ep[e]);
      #pragma unroll
      for (int e = 8; e < 16; ++e) a.u[e] = 0;
    }
    // ---- 8 N-tiles ----
    #pragma unroll
    for (int nt = 0; nt < 8; ++nt) {
      BF16x16 b;
      const unsigned short* bp = wattfrag + ((size_t)(nt * KT_ATT + kt) * 32 + lane) * 16;
      b.q[0] = *(const uint4*)(bp);
      b.q[1] = *(const uint4*)(bp + 8);
      acc[nt] = __builtin_amdgcn_wmma_f32_16x16x32_bf16(
          false, a.v, false, b.v, (short)0, acc[nt], false, false);
    }
  }

  // ---- bias + leaky ReLU, stage alpha tile to LDS ----
  const int c0 = lane & 15;
  #pragma unroll
  for (int nt = 0; nt < 8; ++nt) {
    const int col = nt * 16 + c0;
    const float bv = batt[col];
    #pragma unroll
    for (int v = 0; v < 8; ++v) {
      float av = acc[nt][v] + bv;
      av = (av > 0.0f) ? av : NEG_SLOPE * av;
      s_alpha[wave][v + (hi ? 8 : 0)][col] = av;
    }
  }

  // ---- messages + scatter: each lane owns 4 contiguous cols for all 16 edges
  const int cs = lane * 4;
  #pragma unroll 4
  for (int mm = 0; mm < 16; ++mm) {
    const int src = s_col[wave][mm];
    const int dst = s_row[wave][mm];
    const float4 hv = *(const float4*)(h + (size_t)src * HID + cs);
    const float4 av = *(const float4*)(&s_alpha[wave][mm][cs]);
    float* op = out + (size_t)dst * HID + cs;
    __hip_atomic_fetch_add(op + 0, hv.x * av.x, __ATOMIC_RELAXED, __HIP_MEMORY_SCOPE_AGENT);
    __hip_atomic_fetch_add(op + 1, hv.y * av.y, __ATOMIC_RELAXED, __HIP_MEMORY_SCOPE_AGENT);
    __hip_atomic_fetch_add(op + 2, hv.z * av.z, __ATOMIC_RELAXED, __HIP_MEMORY_SCOPE_AGENT);
    __hip_atomic_fetch_add(op + 3, hv.w * av.w, __ATOMIC_RELAXED, __HIP_MEMORY_SCOPE_AGENT);
  }
}

// ---------------------------------------------------------------------------
extern "C" void kernel_launch(void* const* d_in, const int* in_sizes, int n_in,
                              void* d_out, int out_size, void* d_ws, size_t ws_size,
                              hipStream_t stream) {
  (void)in_sizes; (void)n_in; (void)out_size; (void)ws_size;
  const float*     x     = (const float*)d_in[0];
  const long long* ei    = (const long long*)d_in[1];   // int64 [2, E]
  const float*     eattr = (const float*)d_in[2];
  const float*     Wfc   = (const float*)d_in[3];
  const float*     bfc   = (const float*)d_in[4];
  const float*     Watt  = (const float*)d_in[5];
  const float*     batt  = (const float*)d_in[6];
  float*           out   = (float*)d_out;

  // workspace carve-up (256B aligned segments)
  char*  ws  = (char*)d_ws;
  size_t off = 0;
  float* h = (float*)(ws + off);              off += (size_t)NNODES * HID * 4;   // 51.2 MB
  unsigned short* hbf = (unsigned short*)(ws + off); off += (size_t)NNODES * HID * 2; // 25.6 MB
  float* wfcf = (float*)(ws + off);           off += (size_t)8 * 32 * 32 * 2 * 4; // 64 KB
  unsigned short* wattf = (unsigned short*)(ws + off);
  off += (size_t)8 * KT_ATT * 32 * 16 * 2;                                        // 72 KB

  const int outN = NNODES * HID;
  zero_f32<<<(outN + 1023) / 1024, 1024, 0, stream>>>(out, outN);
  pack_wfc<<<(8 * 32 * 32 + 255) / 256, 256, 0, stream>>>(Wfc, wfcf);
  pack_watt<<<(8 * KT_ATT * 32 * 16 + 255) / 256, 256, 0, stream>>>(Watt, wattf);

  node_gemm<<<(NNODES / 16 + 7) / 8, 256, 0, stream>>>(x, wfcf, bfc, h, hbf);

  edge_gat<<<(NEDGES / 16) / WPB, 32 * WPB, 0, stream>>>(
      ei, eattr, wattf, batt, h, hbf, out);
}
